// Head_36996848288169
// MI455X (gfx1250) — matrix operand order repaired and checked by reading, not compiled, not run on hardware
//
#include <hip/hip_runtime.h>
#include <hip/hip_bf16.h>
#include <stdint.h>

// CDNA5 (gfx1250) wave32 WMMA types
typedef __attribute__((ext_vector_type(16))) __bf16 v16bf;
typedef __attribute__((ext_vector_type(8)))  float  v8f;

union Frag {
    uint32_t u[8];
    uint4    q[2];   // for 128-bit LDS moves
    v16bf    v;
};

// Packed f32x2 -> bf16x2 (RNE); lowers to v_cvt_pk/v_perm on gfx1250.
__device__ __forceinline__ uint32_t pack2_bf16(float lo, float hi) {
    union { __hip_bfloat162 h; uint32_t u; } cv;
    cv.h = __float22bfloat162_rn(make_float2(lo, hi));
    return cv.u;
}
__device__ __forceinline__ uint16_t bf16_1(float x) {
    union { __hip_bfloat16 h; uint16_t u; } cv;
    cv.h = __float2bfloat16(x);
    return cv.u;
}

// Problem constants
constexpr int Tn = 256, Cn = 384, HSn = 64;

// LDS layout (bytes). Strides padded to avoid 64-bank conflicts
// (stride bytes not ≡ 0 mod 256 so 16 lanes land on 16 distinct banks).
constexpr int KSTR = 72;        // Kb row stride (u16), 144 B
constexpr int VSTR = Tn + 8;    // Vt row stride (u16), 528 B
constexpr int QSTR = 72;        // per-wave Q scratch stride (u16), 144 B
constexpr int PSTR = 40;        // per-wave P scratch stride (u16), 80 B
constexpr int WFRAG_U32 = 256;  // one B-fragment: 32 lanes x 8 u32 = 1 KB
constexpr int NWFRAG   = 3 * 12 * 4;  // {Wq,Wk,Wv} x c-chunks x h-tiles = 144

constexpr int OFF_WF = 0;
constexpr int OFF_KB = OFF_WF + NWFRAG * WFRAG_U32 * 4;     // 147456
constexpr int OFF_VT = OFF_KB + Tn * KSTR * 2;              // +36864
constexpr int OFF_SC = OFF_VT + HSn * VSTR * 2;             // +33792
constexpr int SMEM_BYTES = OFF_SC + 8 * (32 * QSTR * 2);    // +36864 = 254976

// Fused causal attention head: one workgroup per batch, 8 waves x 32 q-rows.
__global__ void __launch_bounds__(256, 1) attn_head_fused(
    const float* __restrict__ X,   // [B][T][C]
    const float* __restrict__ Wq,  // [C][HS]
    const float* __restrict__ Wk,
    const float* __restrict__ Wv,
    float* __restrict__ Out)       // [B][T][HS]
{
    extern __shared__ char smem[];
    uint32_t* Wf = reinterpret_cast<uint32_t*>(smem + OFF_WF); // 144 B-frags, bf16
    uint16_t* Kb = reinterpret_cast<uint16_t*>(smem + OFF_KB); // [T][KSTR]
    uint16_t* Vt = reinterpret_cast<uint16_t*>(smem + OFF_VT); // [HS][VSTR] transposed
    uint16_t* Sc = reinterpret_cast<uint16_t*>(smem + OFF_SC); // 8 x [32][QSTR]

    const int b    = blockIdx.x;
    const int w    = threadIdx.x >> 5;   // wave 0..7
    const int lane = threadIdx.x & 31;
    const int n16  = lane & 15;
    const int half = lane >> 4;

    uint16_t* scr = Sc + w * (32 * QSTR);
    const float* Xb = X + (size_t)b * Tn * Cn;
    const float* Wmat[3] = {Wq, Wk, Wv};

    // ---- Phase 0: build all W B-fragments once per block into LDS --------
    // frag f = (p*12 + cc)*4 + ct ; per-lane 8 u32 stored as 2x b128.
    for (int f = w; f < NWFRAG; f += 8) {
        const int p  = f / 48;
        const int cc = (f % 48) / 4;
        const int ct = f & 3;
        const int c0 = cc * 32;
        const float* Wp = Wmat[p];
        Frag frag;
        #pragma unroll
        for (int j = 0; j < 8; ++j) {
            int k = (j < 4) ? (8 * half + 2 * j) : (16 + 8 * half + 2 * (j - 4));
            const float* wp = Wp + (size_t)(c0 + k) * HSn + 16 * ct + n16;
            frag.u[j] = pack2_bf16(wp[0], wp[HSn]);
        }
        uint4* dst = reinterpret_cast<uint4*>(Wf + f * WFRAG_U32 + lane * 8); // 32B-aligned
        dst[0] = frag.q[0];
        dst[1] = frag.q[1];
    }
    __syncthreads();

    // ---- Phase 1: Q/K/V projections for rows [32w, 32w+32), X read once --
    v8f zero = {};
    v8f acc[3][2][4];   // {Q,K,V} x rowtile x h-tile(16)
    #pragma unroll
    for (int p = 0; p < 3; ++p)
        #pragma unroll
        for (int r = 0; r < 2; ++r)
            #pragma unroll
            for (int ct = 0; ct < 4; ++ct) acc[p][r][ct] = zero;

    for (int cc = 0; cc < 12; ++cc) {
        const int c0 = cc * 32;
        Frag xa[2];                                  // X A-fragments (16x32 bf16)
        #pragma unroll
        for (int r = 0; r < 2; ++r) {
            const float* xrow = Xb + (size_t)(32 * w + 16 * r + n16) * Cn + c0;
            #pragma unroll
            for (int j = 0; j < 8; ++j) {
                int k = (j < 4) ? (8 * half + 2 * j) : (16 + 8 * half + 2 * (j - 4));
                float2 xv = *reinterpret_cast<const float2*>(xrow + k); // b64
                xa[r].u[j] = pack2_bf16(xv.x, xv.y);
            }
        }
        #pragma unroll
        for (int p = 0; p < 3; ++p) {
            #pragma unroll
            for (int ct = 0; ct < 4; ++ct) {
                const uint4* src = reinterpret_cast<const uint4*>(
                    Wf + ((p * 12 + cc) * 4 + ct) * WFRAG_U32 + lane * 8);
                Frag bf;
                bf.q[0] = src[0];                    // ds_load_b128
                bf.q[1] = src[1];                    // ds_load_b128
                #pragma unroll
                for (int r = 0; r < 2; ++r)
                    acc[p][r][ct] = __builtin_amdgcn_wmma_f32_16x16x32_bf16(
                        false, xa[r].v, false, bf.v, (short)0, acc[p][r][ct], false, false);
            }
        }
    }

    // K -> Kb (row-major), V -> Vt (transposed), Q -> per-wave scratch.
    #pragma unroll
    for (int r = 0; r < 2; ++r)
        #pragma unroll
        for (int ct = 0; ct < 4; ++ct)
            #pragma unroll
            for (int j = 0; j < 8; ++j) {
                int rowl = 16 * r + j + 8 * half;    // 0..31 within wave slice
                int col  = 16 * ct + n16;            // 0..63 (h)
                int rowg = 32 * w + rowl;            // 0..255 (t)
                Kb[rowg * KSTR + col]  = bf16_1(acc[1][r][ct][j]);
                Vt[col * VSTR + rowg]  = bf16_1(acc[2][r][ct][j]);
                scr[rowl * QSTR + col] = bf16_1(acc[0][r][ct][j]);
            }
    __syncthreads();

    // Reload Q as bf16 A-fragments; bf16 pairs are contiguous u32s.
    Frag qa[2][2];
    #pragma unroll
    for (int r = 0; r < 2; ++r) {
        const uint32_t* qrow = reinterpret_cast<const uint32_t*>(scr + (16 * r + n16) * QSTR);
        #pragma unroll
        for (int hc = 0; hc < 2; ++hc)
            #pragma unroll
            for (int j = 0; j < 8; ++j) {
                int k = (j < 4) ? (8 * half + 2 * j) : (16 + 8 * half + 2 * (j - 4));
                qa[r][hc].u[j] = qrow[16 * hc + (k >> 1)];
            }
    }

    // ---- Phase 2: causal flash attention over K/V ------------------------
    float mi[2][8], li[2][8];
    v8f O[2][4];
    #pragma unroll
    for (int r = 0; r < 2; ++r) {
        #pragma unroll
        for (int j = 0; j < 8; ++j) { mi[r][j] = -__builtin_inff(); li[r][j] = 0.0f; }
        #pragma unroll
        for (int ct = 0; ct < 4; ++ct) O[r][ct] = zero;
    }

    const float sscale = 0.05103103630798288f;   // 384^-0.5 (ref scales by C)

    for (int kc = 0; kc <= w; ++kc) {            // causal: only k <= q chunks
        const int k0 = 32 * kc;

        // K^T B-fragments: element (h, kcol) = K[kcol][h]; h-pairs contiguous.
        Frag kf[2][2];
        #pragma unroll
        for (int t = 0; t < 2; ++t) {
            const uint32_t* krow =
                reinterpret_cast<const uint32_t*>(Kb + (k0 + 16 * t + n16) * KSTR);
            #pragma unroll
            for (int hc = 0; hc < 2; ++hc)
                #pragma unroll
                for (int j = 0; j < 8; ++j)
                    kf[t][hc].u[j] = krow[16 * hc + ((j < 4) ? (4 * half + j) : (8 + 4 * half + (j - 4)))];
        }

        v8f S[2][2];
        #pragma unroll
        for (int r = 0; r < 2; ++r)
            #pragma unroll
            for (int t = 0; t < 2; ++t) {
                v8f c = zero;
                c = __builtin_amdgcn_wmma_f32_16x16x32_bf16(
                        false, qa[r][0].v, false, kf[t][0].v, (short)0, c, false, false);
                c = __builtin_amdgcn_wmma_f32_16x16x32_bf16(
                        false, qa[r][1].v, false, kf[t][1].v, (short)0, c, false, false);
                S[r][t] = c;
            }

        // Scale + causal mask (mask only bites in the diagonal chunk kc==w).
        #pragma unroll
        for (int r = 0; r < 2; ++r)
            #pragma unroll
            for (int t = 0; t < 2; ++t)
                #pragma unroll
                for (int j = 0; j < 8; ++j) {
                    float s  = S[r][t][j] * sscale;
                    int qrow = 32 * w + 16 * r + j + 8 * half;
                    int kcol = k0 + 16 * t + n16;
                    S[r][t][j] = (kcol > qrow) ? -__builtin_inff() : s;
                }

        // Online softmax: each row lives in a 16-lane half-group (wave32).
        #pragma unroll
        for (int r = 0; r < 2; ++r)
            #pragma unroll
            for (int j = 0; j < 8; ++j) {
                float rowm = fmaxf(S[r][0][j], S[r][1][j]);
                #pragma unroll
                for (int m = 8; m >= 1; m >>= 1)
                    rowm = fmaxf(rowm, __shfl_xor(rowm, m, 32));
                float newm = fmaxf(mi[r][j], rowm);
                float corr = __expf(mi[r][j] - newm);
                float e0 = __expf(S[r][0][j] - newm);
                float e1 = __expf(S[r][1][j] - newm);
                S[r][0][j] = e0; S[r][1][j] = e1;
                float rs = e0 + e1;
                #pragma unroll
                for (int m = 8; m >= 1; m >>= 1)
                    rs += __shfl_xor(rs, m, 32);
                li[r][j] = li[r][j] * corr + rs;
                mi[r][j] = newm;
                #pragma unroll
                for (int ct = 0; ct < 4; ++ct) O[r][ct][j] *= corr;
            }

        // P: D-layout -> A-layout via per-wave LDS bounce (same-wave LDS is
        // in-order; wait + clobber stop compiler reordering).
        #pragma unroll
        for (int r = 0; r < 2; ++r)
            #pragma unroll
            for (int t = 0; t < 2; ++t)
                #pragma unroll
                for (int j = 0; j < 8; ++j)
                    scr[(16 * r + j + 8 * half) * PSTR + 16 * t + n16] =
                        bf16_1(S[r][t][j]);
        asm volatile("s_wait_dscnt 0" ::: "memory");

        Frag pa[2];
        #pragma unroll
        for (int r = 0; r < 2; ++r) {
            const uint32_t* prow =
                reinterpret_cast<const uint32_t*>(scr + (16 * r + n16) * PSTR);
            #pragma unroll
            for (int j = 0; j < 8; ++j) {
                int k = (j < 4) ? (8 * half + 2 * j) : (16 + 8 * half + 2 * (j - 4));
                pa[r].u[j] = prow[k >> 1];
            }
        }

        // O += P @ V  (V transposed in LDS: k-pairs are contiguous u32s).
        #pragma unroll
        for (int ct = 0; ct < 4; ++ct) {
            const uint32_t* vcol =
                reinterpret_cast<const uint32_t*>(Vt + (16 * ct + n16) * VSTR);
            Frag vf;
            #pragma unroll
            for (int j = 0; j < 8; ++j)
                vf.u[j] = vcol[(k0 >> 1) + ((j < 4) ? (4 * half + j) : (8 + 4 * half + (j - 4)))];
            #pragma unroll
            for (int r = 0; r < 2; ++r)
                O[r][ct] = __builtin_amdgcn_wmma_f32_16x16x32_bf16(
                    false, pa[r].v, false, vf.v, (short)0, O[r][ct], false, false);
        }
        asm volatile("" ::: "memory");
    }

    // ---- epilogue: normalize & store (64B contiguous per half-wave) ------
    float* Ob = Out + (size_t)b * Tn * HSn;
    #pragma unroll
    for (int r = 0; r < 2; ++r)
        #pragma unroll
        for (int ct = 0; ct < 4; ++ct)
            #pragma unroll
            for (int j = 0; j < 8; ++j) {
                int rowg = 32 * w + 16 * r + j + 8 * half;
                int col  = 16 * ct + n16;
                Ob[rowg * HSn + col] = O[r][ct][j] / li[r][j];
            }
}

extern "C" void kernel_launch(void* const* d_in, const int* in_sizes, int n_in,
                              void* d_out, int out_size, void* d_ws, size_t ws_size,
                              hipStream_t stream) {
    (void)in_sizes; (void)n_in; (void)out_size; (void)d_ws; (void)ws_size;
    const float* X  = (const float*)d_in[0];
    const float* Wq = (const float*)d_in[1];
    const float* Wk = (const float*)d_in[2];
    const float* Wv = (const float*)d_in[3];
    float* Out = (float*)d_out;

    hipFuncSetAttribute(reinterpret_cast<const void*>(&attn_head_fused),
                        hipFuncAttributeMaxDynamicSharedMemorySize, SMEM_BYTES);

    attn_head_fused<<<dim3(256), dim3(256), SMEM_BYTES, stream>>>(X, Wq, Wk, Wv, Out);
}
